// STEB_UNet_68083821576651
// MI455X (gfx1250) — matrix operand order repaired and checked by reading, not compiled, not run on hardware
//
#include <hip/hip_runtime.h>
#include <math.h>

// ---------------------------------------------------------------------------
// Swin block pair (W-MSA + SW-MSA) for MI455X / gfx1250, bf16 WMMA everywhere.
// B=4, H=W=256, C=96, WS=8, NH=3, head_dim=32, HID=384.
// All WMMA fragment loads are vectorized 16B/32B per lane (ds_load_b128 /
// global_load_b128) using a contiguous K = half*16 + e lane order in A and B; weights are
// pre-swizzled into fragment-major order so B operands are single wide loads.
// ---------------------------------------------------------------------------

typedef unsigned short u16;
typedef __attribute__((ext_vector_type(16))) __bf16 v16bf;
typedef __attribute__((ext_vector_type(8)))  __bf16 v8bf;
typedef __attribute__((ext_vector_type(8)))  float  v8f;

#define BATCH 4
#define HW    256
#define CH    96
#define NTOK  64          // tokens per 8x8 window
#define QKVC  288
#define HID   384
#define NWIN  4096        // B * (256/8)^2
#define SHIFT_ 4

// Native f32 -> bf16 convert (hardware cvt on gfx1250).
__device__ __forceinline__ u16 f2bf(float f) {
    return __builtin_bit_cast(u16, static_cast<__bf16>(f));
}

// ---- WMMA helpers (A/B: contiguous K = half*16 + e per lane, wave32) ----------------------

__device__ __forceinline__ v8f wmma_bf16(v16bf a, v16bf b, v8f c) {
    // (neg_a, A, neg_b, B, c_mod, C, reuse_a, reuse_b)
    return __builtin_amdgcn_wmma_f32_16x16x32_bf16(false, a, false, b,
                                                   (short)0, c, false, false);
}

// A fragment 16x32 (MxK) from row-major bf16, leading dim ld (elements).
// Per lane: two contiguous 16B chunks at K = half*16 and K = half*16 + 8.
__device__ __forceinline__ v16bf load_a(const u16* src, int ld, int row0,
                                        int col0, int lane) {
    int m = lane & 15, half = lane >> 4;
    const u16* p = src + (row0 + m) * ld + col0 + half * 16;
    v8bf lo = *(const v8bf*)p;
    v8bf hi = *(const v8bf*)(p + 8);
    return __builtin_shufflevector(lo, hi, 0, 1, 2, 3, 4, 5, 6, 7,
                                   8, 9, 10, 11, 12, 13, 14, 15);
}

// B fragment 32x16 (KxN) where the source is stored N-major (row = N index,
// 16 consecutive K elements per lane): logical B(k,n) = src[(nrow0+n)*ld +
// kcol0 + k]. One contiguous 32B load per lane.
__device__ __forceinline__ v16bf load_bt(const u16* src, int ld, int nrow0,
                                         int kcol0, int lane) {
    int n = lane & 15, half = lane >> 4;
    const u16* p = src + (nrow0 + n) * ld + kcol0 + half * 16;
    return *(const v16bf*)p;
}

// B fragment from fragment-major pre-swizzled weights: tile (kt,nt) holds
// 512 bf16, lane-major (16 contiguous per lane). One 32B load per lane.
__device__ __forceinline__ v16bf load_bw(const u16* w, int ntiles, int kt,
                                         int nt, int lane) {
    const u16* p = w + ((kt * ntiles + nt) << 9) + (lane << 4);
    return *(const v16bf*)p;
}

// C/D f32 fragment layout: lane 0-15 -> N=lane, reg r -> M=r;
// lane 16-31 -> N=lane-16, reg r -> M=8+r.
__device__ __forceinline__ void store_c_bf16(u16* dst, int ld, int row0,
                                             int col0, int lane, v8f c) {
    int n = lane & 15, mh = (lane >> 4) * 8;
#pragma unroll
    for (int r = 0; r < 8; ++r)
        dst[(row0 + mh + r) * ld + col0 + n] = f2bf(c[r]);
}

__device__ __forceinline__ int shift_region(int g) {
    // slices on the rolled grid: [0,248)->0, [248,252)->1, [252,256)->2
    return (g < HW - 8) ? 0 : ((g < HW - SHIFT_) ? 1 : 2);
}

// ---------------------------------------------------------------------------
// Weight fp32 -> bf16 conversion + fragment-major swizzle.
// Source: row-major [K x N]. Dest tile (kt,nt) 32x16: lane l = half*16 + n,
// element e -> src (kt*32 + half*16 + e, nt*16 + n).
// ---------------------------------------------------------------------------
__global__ void swin_cvt_swizzle(const float* __restrict__ src,
                                 u16* __restrict__ dst, int K, int N) {
    int i = blockIdx.x * blockDim.x + threadIdx.x;
    if (i >= K * N) return;
    int tile = i >> 9, idx = i & 511;
    int lane = idx >> 4, e = idx & 15;
    int n = lane & 15, half = lane >> 4;
    int ntiles = N >> 4;
    int kt = tile / ntiles, nt = tile - kt * ntiles;
    dst[i] = f2bf(src[(kt * 32 + half * 16 + e) * N + nt * 16 + n]);
}

// ---------------------------------------------------------------------------
// Fused LN1 + window attention + proj + residual. One window per block.
// In-place on x. Shift handled via rolled index arithmetic.
// ---------------------------------------------------------------------------
__global__ __launch_bounds__(256) void swin_attn_kernel(
    float* __restrict__ x,            // [B, HW*HW, CH] in/out
    const float* __restrict__ ln_g,   // [CH]
    const float* __restrict__ ln_b,   // [CH]
    const u16*   __restrict__ wqkv,   // [96x288] bf16, fragment-major
    const float* __restrict__ bqkv,   // [QKVC]
    const float* __restrict__ rpb,    // [225, 3]
    const u16*   __restrict__ wproj,  // [96x96] bf16, fragment-major
    const float* __restrict__ bproj,  // [CH]
    int shift)                        // 0 or SHIFT_
{
    extern __shared__ char smem[];
    float* Xo  = (float*)(smem);                 // [64][96]   24576 B
    u16*   Xn  = (u16*)  (smem + 24576);         // [64][96]   12288 B
    u16*   QK  = (u16*)  (smem + 36864);         // [64][192]  24576 B (Q|K)
    u16*   VT  = (u16*)  (smem + 61440);         // [96][64]   12288 B (V^T)
    float* S   = (float*)(smem + 73728);         // [3][64][64] 49152 B
    u16*   P   = (u16*)  (smem + 122880);        // [3][64][64] 24576 B
    u16*   O   = (u16*)  (smem + 147456);        // [64][96]   12288 B

    const int wid = blockIdx.x;
    const int bb  = wid >> 10;          // batch
    const int wi  = wid & 1023;
    const int wr  = wi >> 5, wc = wi & 31;
    const int tid = threadIdx.x;
    // wave id is wave-uniform: make it scalar so tile loops use SALU control.
    const int wave = __builtin_amdgcn_readfirstlane(tid >> 5);
    const int lane = tid & 31;
    const long long xbase = (long long)bb * (HW * HW) * CH;

    if (tid < 54) __builtin_prefetch(wqkv + tid * 512, 0, 3);  // 54 weight tiles

    // ---- 1) load window (rolled coords) into LDS, float4 ------------------
    for (int e = tid; e < NTOK * (CH / 4); e += 256) {
        int t = e / 24, c4 = e - t * 24;
        int i = t >> 3, j = t & 7;
        int gh = (wr * 8 + i + shift) & (HW - 1);
        int gw = (wc * 8 + j + shift) & (HW - 1);
        ((float4*)Xo)[e] =
            *(const float4*)(x + xbase + (long long)(gh * HW + gw) * CH + c4 * 4);
    }
    __syncthreads();

    // ---- 2) LayerNorm per token ------------------------------------------
    if (tid < NTOK) {
        const float* row = Xo + tid * CH;
        float s = 0.f;
#pragma unroll 8
        for (int c = 0; c < CH; ++c) s += row[c];
        float mu = s * (1.f / CH);
        float ss = 0.f;
#pragma unroll 8
        for (int c = 0; c < CH; ++c) { float d = row[c] - mu; ss += d * d; }
        float inv = rsqrtf(ss * (1.f / CH) + 1e-5f);
        for (int c = 0; c < CH; ++c)
            Xn[tid * CH + c] = f2bf((row[c] - mu) * inv * ln_g[c] + ln_b[c]);
    }
    __syncthreads();

    // ---- 3) QKV = Xn @ Wqkv + b  (64x96 @ 96x288) -------------------------
    //   Q (cols 0..95, pre-scaled), K (96..191) -> QK row-major [64][192]
    //   V (192..287) -> VT [96][64] transposed
    const float scale = 0.17677669529663687f;  // 32^-0.5
    for (int t = wave; t < 4 * 18; t += 8) {
        int mt = t / 18, nt = t - mt * 18;
        v8f acc;
        float bias = bqkv[nt * 16 + (lane & 15)];
#pragma unroll
        for (int r = 0; r < 8; ++r) acc[r] = bias;
#pragma unroll
        for (int kt = 0; kt < 3; ++kt) {
            v16bf af = load_a(Xn, CH, mt * 16, kt * 32, lane);
            v16bf bf = load_bw(wqkv, 18, kt, nt, lane);
            acc = wmma_bf16(af, bf, acc);
        }
        if (nt < 6) {               // Q: fold softmax scale
#pragma unroll
            for (int r = 0; r < 8; ++r) acc[r] *= scale;
            store_c_bf16(QK, 192, mt * 16, nt * 16, lane, acc);
        } else if (nt < 12) {       // K
            store_c_bf16(QK, 192, mt * 16, nt * 16, lane, acc);
        } else {                    // V -> transposed
            int n0 = lane & 15, mh = (lane >> 4) * 8;
#pragma unroll
            for (int r = 0; r < 8; ++r)
                VT[((nt - 12) * 16 + n0) * NTOK + mt * 16 + mh + r] = f2bf(acc[r]);
        }
    }
    __syncthreads();

    // ---- 4) S = Q K^T + rel-pos bias (+ shift mask) -----------------------
    for (int t = wave; t < 3 * 4 * 4; t += 8) {
        int h = t >> 4, mt = (t >> 2) & 3, nt = t & 3;
        v8f acc;
#pragma unroll
        for (int r = 0; r < 8; ++r) acc[r] = 0.f;
        v16bf af = load_a(QK, 192, mt * 16, h * 32, lane);            // Q
        v16bf bf = load_bt(QK, 192, nt * 16, CH + h * 32, lane);      // K^T
        acc = wmma_bf16(af, bf, acc);
        int n0 = lane & 15, mh = (lane >> 4) * 8;
#pragma unroll
        for (int r = 0; r < 8; ++r) {
            int m = mt * 16 + mh + r;
            int n = nt * 16 + n0;
            int mi = m >> 3, mj = m & 7, ni = n >> 3, nj = n & 7;
            int ridx = (mi - ni + 7) * 15 + (mj - nj + 7);
            float v = acc[r] + rpb[ridx * 3 + h];
            if (shift) {
                int rm = shift_region(wr * 8 + mi) * 3 + shift_region(wc * 8 + mj);
                int rn = shift_region(wr * 8 + ni) * 3 + shift_region(wc * 8 + nj);
                if (rm != rn) v -= 100.f;
            }
            S[(h * NTOK + m) * NTOK + n] = v;
        }
    }
    __syncthreads();

    // ---- 5) softmax over rows (192 rows) ----------------------------------
    if (tid < 3 * NTOK) {
        float* row = S + tid * NTOK;
        float mx = row[0];
#pragma unroll 8
        for (int n = 1; n < NTOK; ++n) mx = fmaxf(mx, row[n]);
        float sum = 0.f;
#pragma unroll 8
        for (int n = 0; n < NTOK; ++n) { float e = __expf(row[n] - mx); row[n] = e; sum += e; }
        float inv = 1.f / sum;
        for (int n = 0; n < NTOK; ++n) P[tid * NTOK + n] = f2bf(row[n] * inv);
    }
    __syncthreads();

    // ---- 6) O = P @ V  (per head: 64x64 @ 64x32), V read via VT ----------
    for (int t = wave; t < 3 * 4 * 2; t += 8) {
        int h = t >> 3, mt = (t >> 1) & 3, nt = t & 1;
        v8f acc;
#pragma unroll
        for (int r = 0; r < 8; ++r) acc[r] = 0.f;
#pragma unroll
        for (int kt = 0; kt < 2; ++kt) {
            v16bf af = load_a(P + h * NTOK * NTOK, NTOK, mt * 16, kt * 32, lane);
            v16bf bf = load_bt(VT, NTOK, h * 32 + nt * 16, kt * 32, lane);
            acc = wmma_bf16(af, bf, acc);
        }
        store_c_bf16(O, CH, mt * 16, h * 32 + nt * 16, lane, acc);
    }
    __syncthreads();

    // ---- 7) out = Xo + O @ Wproj + b  -> global (same rolled positions) ---
    for (int t = wave; t < 4 * 6; t += 8) {
        int mt = t / 6, nt = t - mt * 6;
        v8f acc;
        float bias = bproj[nt * 16 + (lane & 15)];
#pragma unroll
        for (int r = 0; r < 8; ++r) acc[r] = bias;
#pragma unroll
        for (int kt = 0; kt < 3; ++kt) {
            v16bf af = load_a(O, CH, mt * 16, kt * 32, lane);
            v16bf bf = load_bw(wproj, 6, kt, nt, lane);
            acc = wmma_bf16(af, bf, acc);
        }
        int n0 = lane & 15, mh = (lane >> 4) * 8;
#pragma unroll
        for (int r = 0; r < 8; ++r) {
            int m = mt * 16 + mh + r;
            int c = nt * 16 + n0;
            int i = m >> 3, j = m & 7;
            int gh = (wr * 8 + i + shift) & (HW - 1);
            int gw = (wc * 8 + j + shift) & (HW - 1);
            x[xbase + (long long)(gh * HW + gw) * CH + c] = Xo[m * CH + c] + acc[r];
        }
    }
}

// ---------------------------------------------------------------------------
// Fused LN2 + MLP (96 -> 384 -> 96, exact GELU) + residual.
// 64 tokens per block, in-place on x.
// ---------------------------------------------------------------------------
__global__ __launch_bounds__(256) void swin_mlp_kernel(
    float* __restrict__ x,
    const float* __restrict__ ln_g, const float* __restrict__ ln_b,
    const u16*   __restrict__ w1,   const float* __restrict__ b1,
    const u16*   __restrict__ w2,   const float* __restrict__ b2)
{
    extern __shared__ char smem[];
    float* Xo = (float*)(smem);            // [64][96]  24576 B
    u16*   Xn = (u16*)  (smem + 24576);    // [64][96]  12288 B
    u16*   H1 = (u16*)  (smem + 36864);    // [64][384] 49152 B

    const int tid = threadIdx.x;
    const int wave = __builtin_amdgcn_readfirstlane(tid >> 5);
    const int lane = tid & 31;
    const long long tok0 = (long long)blockIdx.x * NTOK;

    if (tid < 72) __builtin_prefetch(w1 + tid * 512, 0, 3);  // 72 fc1 tiles

    for (int e = tid; e < NTOK * (CH / 4); e += 256)
        ((float4*)Xo)[e] = *(const float4*)(x + tok0 * CH + e * 4);
    __syncthreads();

    if (tid < NTOK) {
        const float* row = Xo + tid * CH;
        float s = 0.f;
#pragma unroll 8
        for (int c = 0; c < CH; ++c) s += row[c];
        float mu = s * (1.f / CH);
        float ss = 0.f;
#pragma unroll 8
        for (int c = 0; c < CH; ++c) { float d = row[c] - mu; ss += d * d; }
        float inv = rsqrtf(ss * (1.f / CH) + 1e-5f);
        for (int c = 0; c < CH; ++c)
            Xn[tid * CH + c] = f2bf((row[c] - mu) * inv * ln_g[c] + ln_b[c]);
    }
    __syncthreads();

    // fc1 + GELU: 64x96 @ 96x384
    for (int t = wave; t < 4 * 24; t += 8) {
        int mt = t / 24, nt = t - mt * 24;
        v8f acc;
        float bias = b1[nt * 16 + (lane & 15)];
#pragma unroll
        for (int r = 0; r < 8; ++r) acc[r] = bias;
#pragma unroll
        for (int kt = 0; kt < 3; ++kt) {
            v16bf af = load_a(Xn, CH, mt * 16, kt * 32, lane);
            v16bf bf = load_bw(w1, 24, kt, nt, lane);
            acc = wmma_bf16(af, bf, acc);
        }
        int n0 = lane & 15, mh = (lane >> 4) * 8;
#pragma unroll
        for (int r = 0; r < 8; ++r) {
            float v = acc[r];
            v = 0.5f * v * (1.0f + erff(v * 0.7071067811865476f));
            H1[(mt * 16 + mh + r) * HID + nt * 16 + n0] = f2bf(v);
        }
    }
    __syncthreads();

    // fc2 + residual: 64x384 @ 384x96
    for (int t = wave; t < 4 * 6; t += 8) {
        int mt = t / 6, nt = t - mt * 6;
        v8f acc;
        float bias = b2[nt * 16 + (lane & 15)];
#pragma unroll
        for (int r = 0; r < 8; ++r) acc[r] = bias;
#pragma unroll
        for (int kt = 0; kt < 12; ++kt) {
            v16bf af = load_a(H1, HID, mt * 16, kt * 32, lane);
            v16bf bf = load_bw(w2, 6, kt, nt, lane);
            acc = wmma_bf16(af, bf, acc);
        }
        int n0 = lane & 15, mh = (lane >> 4) * 8;
#pragma unroll
        for (int r = 0; r < 8; ++r) {
            int m = mt * 16 + mh + r, c = nt * 16 + n0;
            x[(tok0 + m) * CH + c] = Xo[m * CH + c] + acc[r];
        }
    }
}

// ---------------------------------------------------------------------------
// Host side
// ---------------------------------------------------------------------------
extern "C" void kernel_launch(void* const* d_in, const int* in_sizes, int n_in,
                              void* d_out, int out_size, void* d_ws, size_t ws_size,
                              hipStream_t stream) {
    (void)in_sizes; (void)n_in; (void)out_size; (void)ws_size;

    const float* x_in   = (const float*)d_in[0];
    const float* ln1_g  = (const float*)d_in[1];
    const float* ln1_b  = (const float*)d_in[2];
    const float* qkv_w  = (const float*)d_in[3];
    const float* qkv_b  = (const float*)d_in[4];
    const float* rpb    = (const float*)d_in[5];
    const float* proj_w = (const float*)d_in[6];
    const float* proj_b = (const float*)d_in[7];
    const float* ln2_g  = (const float*)d_in[8];
    const float* ln2_b  = (const float*)d_in[9];
    const float* fc1_w  = (const float*)d_in[10];
    const float* fc1_b  = (const float*)d_in[11];
    const float* fc2_w  = (const float*)d_in[12];
    const float* fc2_b  = (const float*)d_in[13];

    float* xbuf = (float*)d_out;
    u16*   wbf  = (u16*)d_ws;

    // Allow large dynamic LDS (320 KB WGP pool on CDNA5).
    static const size_t ATTN_SMEM = 159744;
    static const size_t MLP_SMEM  = 86016;
    (void)hipFuncSetAttribute((const void*)swin_attn_kernel,
                              hipFuncAttributeMaxDynamicSharedMemorySize, (int)ATTN_SMEM);
    (void)hipFuncSetAttribute((const void*)swin_mlp_kernel,
                              hipFuncAttributeMaxDynamicSharedMemorySize, (int)MLP_SMEM);

    // x -> out (all subsequent kernels operate in place on d_out)
    hipMemcpyAsync(xbuf, x_in, (size_t)BATCH * HW * HW * CH * sizeof(float),
                   hipMemcpyDeviceToDevice, stream);

    // bf16 fragment-major weight staging in d_ws. Per-d layout (u16 elems):
    //   qkv @ +0 (27648), proj @ +27648 (9216), fc1 @ +36864 (36864), fc2 @ +73728 (36864)
    const int PERD = 110592;
    for (int d = 0; d < 2; ++d) {
        u16* base = wbf + d * PERD;
        swin_cvt_swizzle<<<(27648 + 255) / 256, 256, 0, stream>>>(qkv_w  + d * 27648, base,          96, 288);
        swin_cvt_swizzle<<<( 9216 + 255) / 256, 256, 0, stream>>>(proj_w + d *  9216, base + 27648,  96,  96);
        swin_cvt_swizzle<<<(36864 + 255) / 256, 256, 0, stream>>>(fc1_w  + d * 36864, base + 36864,  96, 384);
        swin_cvt_swizzle<<<(36864 + 255) / 256, 256, 0, stream>>>(fc2_w  + d * 36864, base + 73728, 384,  96);
    }

    for (int d = 0; d < 2; ++d) {
        u16* base = wbf + d * PERD;
        int shift = d ? SHIFT_ : 0;
        swin_attn_kernel<<<NWIN, 256, ATTN_SMEM, stream>>>(
            xbuf, ln1_g + d * CH, ln1_b + d * CH,
            base, qkv_b + d * QKVC, rpb + d * 675,
            base + 27648, proj_b + d * CH, shift);
        swin_mlp_kernel<<<NWIN, 256, MLP_SMEM, stream>>>(
            xbuf, ln2_g + d * CH, ln2_b + d * CH,
            base + 36864, fc1_b + d * HID,
            base + 73728, fc2_b + d * CH);
    }
}
